// ALiBiSelfAttention_73083163508887
// MI455X (gfx1250) — matrix operand order repaired and checked by reading, not compiled, not run on hardware
//
#include <hip/hip_runtime.h>
#include <hip/hip_bf16.h>

typedef __attribute__((ext_vector_type(16))) __bf16 v16bf;
typedef __attribute__((ext_vector_type(8)))  float  v8f;
typedef __attribute__((ext_vector_type(4)))  float  v4f;
typedef __attribute__((ext_vector_type(4)))  unsigned int v4u;

union BF16x16 {           // 16 bf16 = 32 bytes
    v16bf v;
    unsigned short s[16];
    v4u u[2];
};
union BF16x8 {            // 8 bf16 = 16 bytes
    unsigned short s[8];
    v4u u;
};

__device__ __forceinline__ unsigned short f2bf(float f) {
    unsigned int u = __float_as_uint(f);
    return (unsigned short)((u + 0x7FFFu + ((u >> 16) & 1u)) >> 16);  // RNE
}

// Low 32 bits of a flat LDS pointer ARE the LDS byte address (ISA 10.2).
__device__ __forceinline__ unsigned lds_addr_of(const void* p) {
    return (unsigned)(size_t)p;
}
// Async DMA: 16B global -> LDS, tracked by ASYNCcnt (gfx1250 path).
__device__ __forceinline__ void async_ld_b128(unsigned lds_dst, const void* gsrc) {
    asm volatile("global_load_async_to_lds_b128 %0, %1, off"
                 :
                 : "v"(lds_dst), "v"(gsrc)
                 : "memory");
}
__device__ __forceinline__ void wait_async0() {
    asm volatile("s_wait_asynccnt 0" ::: "memory");
}

// ---------------------------------------------------------------------------
// One-shot elementwise f32 -> bf16 conversion (n multiple of 2048)
// ---------------------------------------------------------------------------
__global__ __launch_bounds__(256)
void cvt_f32_bf16(const float* __restrict__ in, unsigned short* __restrict__ out) {
    const size_t i = ((size_t)blockIdx.x * 256 + threadIdx.x) * 8;
    v4f f0 = *(const v4f*)(in + i);
    v4f f1 = *(const v4f*)(in + i + 4);
    BF16x8 pk;
    #pragma unroll
    for (int j = 0; j < 4; ++j) {
        pk.s[j]     = f2bf(f0[j]);
        pk.s[4 + j] = f2bf(f1[j]);
    }
    *(v4u*)(out + i) = pk.u;
}

// ---------------------------------------------------------------------------
// bf16 WMMA GEMM: C[M,N] = A[M,K] * B[K,N]   (A,B bf16 in; C bf16 or f32 out)
// Block tile 128x64, 8 waves (256 thr), wave tile 32x32 (2x2 WMMA frags), BK=32
// A tile staged with async global->LDS DMA; B tile transposed through VGPRs.
// ---------------------------------------------------------------------------
#define BM 128
#define BN 64
#define BK 32
#define ASTR 40   // padded LDS strides (elements) to dodge bank conflicts
#define BSTR 40

template <bool OUT_BF16>
__global__ __launch_bounds__(256)
void gemm_bf16(const unsigned short* __restrict__ A,
               const unsigned short* __restrict__ B,
               void* __restrict__ Cv, int M, int N, int K) {
    __shared__ unsigned short As[BM * ASTR];   // [M][K] row-major
    __shared__ unsigned short Bs[BN * BSTR];   // [N][K] (B transposed)

    const int tid  = threadIdx.x;
    const int lane = tid & 31;
    const int wid  = tid >> 5;
    const int wm   = wid >> 1;      // 0..3
    const int wn   = wid & 1;       // 0..1
    const int m0   = blockIdx.y * BM;
    const int n0   = blockIdx.x * BN;
    const int hl   = lane >> 4;     // 0 or 1 (16-lane half)
    const int l16  = lane & 15;

    v8f acc[2][2] = {};

    for (int k0 = 0; k0 < K; k0 += BK) {
        __syncthreads();
        // ---- stage A tile: 128 rows x 32 cols bf16, async DMA (32B / thread)
        {
            const int row  = tid >> 1;
            const int part = tid & 1;
            const unsigned short* src = A + (size_t)(m0 + row) * K + k0 + part * 16;
            if (k0 + BK < K) __builtin_prefetch(src + BK, 0, 3);
            unsigned short* dst = As + row * ASTR + part * 16;
            async_ld_b128(lds_addr_of(dst),     src);
            async_ld_b128(lds_addr_of(dst + 8), src + 8);
        }
        // ---- stage B tile transposed: Bs[n][k] = B[k0+k][n0+n] (8 elems/thread)
        {
            const int kk = tid >> 3;
            const int nb = (tid & 7) * 8;
            const unsigned short* src = B + (size_t)(k0 + kk) * N + n0 + nb;
            if (k0 + BK < K) __builtin_prefetch(src + (size_t)BK * N, 0, 3);
            BF16x8 pk;
            pk.u = *(const v4u*)src;
            #pragma unroll
            for (int j = 0; j < 8; ++j)
                Bs[(nb + j) * BSTR + kk] = pk.s[j];
        }
        wait_async0();
        __syncthreads();

        // ---- fragments + 4 WMMAs
        BF16x16 a[2], bfr[2];
        #pragma unroll
        for (int f = 0; f < 2; ++f) {
            // A 16x32 layout: lane holds row l16, elems {hl*8+0..7, hl*8+16..23}
            const unsigned short* ap = As + (wm * 32 + f * 16 + l16) * ASTR + hl * 8;
            a[f].u[0] = *(const v4u*)ap;
            a[f].u[1] = *(const v4u*)(ap + 16);
            // B 32x16 layout: lane holds col l16, elems {hl*8+0..7, hl*8+16..23}
            const unsigned short* bp = Bs + (wn * 32 + f * 16 + l16) * BSTR + hl * 8;
            bfr[f].u[0] = *(const v4u*)bp;
            bfr[f].u[1] = *(const v4u*)(bp + 16);
        }
        #pragma unroll
        for (int fm = 0; fm < 2; ++fm)
            #pragma unroll
            for (int fn = 0; fn < 2; ++fn)
                acc[fm][fn] = __builtin_amdgcn_wmma_f32_16x16x32_bf16(
                    false, a[fm].v, false, bfr[fn].v, (short)0, acc[fm][fn],
                    false, false);
    }

    // ---- epilogue: C/D layout row = r + 8*hl, col = l16
    #pragma unroll
    for (int fm = 0; fm < 2; ++fm)
        #pragma unroll
        for (int fn = 0; fn < 2; ++fn)
            #pragma unroll
            for (int r = 0; r < 8; ++r) {
                const int row = m0 + wm * 32 + fm * 16 + r + hl * 8;
                const int col = n0 + wn * 32 + fn * 16 + l16;
                if (OUT_BF16)
                    ((unsigned short*)Cv)[(size_t)row * N + col] = f2bf(acc[fm][fn][r]);
                else
                    ((float*)Cv)[(size_t)row * N + col] = acc[fm][fn][r];
            }
}

// ---------------------------------------------------------------------------
// Flash attention with ALiBi + causal mask, bf16 in / bf16 out.
// Grid: B*H*(T/64) blocks, 128 threads (4 waves), wave owns 16 query rows.
// K tile staged by async DMA; V^T transpose-staged through VGPRs.
// ---------------------------------------------------------------------------
#define TSEQ 2048
#define NH   16
#define HD   64
#define KSTR 72
#define VSTR 40
#define PSTR 40

__global__ __launch_bounds__(128)
void attn_alibi(const unsigned short* __restrict__ qkv,
                unsigned short* __restrict__ out) {
    __shared__ unsigned short Ks[32 * KSTR];        // [key][dim] bf16
    __shared__ unsigned short Vt[HD * VSTR];        // [dim][key] bf16
    __shared__ unsigned short Ps[4 * 16 * PSTR];    // per-wave P tile [row][key]

    const int tid  = threadIdx.x;
    const int lane = tid & 31;
    const int wid  = tid >> 5;
    const int hl   = lane >> 4;
    const int l16  = lane & 15;

    const int qtiles = TSEQ / 64;
    const int q0 = (blockIdx.x % qtiles) * 64;
    const int bh = blockIdx.x / qtiles;
    const int b  = bh / NH;
    const int h  = bh % NH;
    const float slope = exp2f(-0.5f * (float)(h + 1));   // 2^(-8/16*(h+1))
    const float scale = 0.125f;                          // 1/sqrt(64)

    const int qr0  = q0 + wid * 16;
    const int qrow = qr0 + l16;

    // ---- Q fragments load straight from bf16 global (no conversion)
    BF16x16 aq[2];
    const unsigned short* qb = qkv + (size_t)(b * TSEQ + qrow) * 3072 + h * HD;
    #pragma unroll
    for (int f = 0; f < 2; ++f) {
        const int d0 = f * 32 + hl * 8;
        aq[f].u[0] = *(const v4u*)(qb + d0);        // dims d0..d0+7
        aq[f].u[1] = *(const v4u*)(qb + d0 + 16);   // dims d0+16..d0+23
    }

    float m_run[8], l_run[8];
    #pragma unroll
    for (int r = 0; r < 8; ++r) { m_run[r] = -1e30f; l_run[r] = 0.f; }
    v8f o[4] = {};

    for (int j0 = 0; j0 < q0 + 64; j0 += 32) {
        __syncthreads();
        // ---- stage K (async DMA) and V^T (transpose) tiles
        {
            const int key = tid >> 2;
            const int d0  = (tid & 3) * 16;
            const unsigned short* kp = qkv + (size_t)(b * TSEQ + j0 + key) * 3072
                                       + 1024 + h * HD + d0;
            unsigned short* kd = Ks + key * KSTR + d0;
            async_ld_b128(lds_addr_of(kd),     kp);
            async_ld_b128(lds_addr_of(kd + 8), kp + 8);
            BF16x16 vv;
            vv.u[0] = *(const v4u*)(kp + 1024);
            vv.u[1] = *(const v4u*)(kp + 1032);
            #pragma unroll
            for (int j = 0; j < 16; ++j)
                Vt[(d0 + j) * VSTR + key] = vv.s[j];
        }
        wait_async0();
        __syncthreads();

        if (j0 > qr0 + 15) continue;   // whole tile causally masked for this wave

        // ---- S = Q K^T (16 queries x 32 keys): 4 WMMAs
        v8f s[2] = {};
        #pragma unroll
        for (int nb = 0; nb < 2; ++nb)
            #pragma unroll
            for (int f = 0; f < 2; ++f) {
                BF16x16 bk;
                const unsigned short* kp = Ks + (nb * 16 + l16) * KSTR
                                           + f * 32 + hl * 8;
                bk.u[0] = *(const v4u*)kp;
                bk.u[1] = *(const v4u*)(kp + 16);
                s[nb] = __builtin_amdgcn_wmma_f32_16x16x32_bf16(
                    false, aq[f].v, false, bk.v, (short)0, s[nb], false, false);
            }

        // ---- ALiBi + causal mask + online softmax (rows live in 16-lane halves)
        float pv[2][8];
        #pragma unroll
        for (int r = 0; r < 8; ++r) {
            const int qr = qr0 + r + hl * 8;
            float best = -1e30f;
            #pragma unroll
            for (int nb = 0; nb < 2; ++nb) {
                const int key = j0 + nb * 16 + l16;
                float v = s[nb][r] * scale + slope * (float)(key - qr);
                if (key > qr) v = -1e30f;
                pv[nb][r] = v;
                best = fmaxf(best, v);
            }
            #pragma unroll
            for (int off = 1; off < 16; off <<= 1)
                best = fmaxf(best, __shfl_xor(best, off, 16));
            const float mnew  = fmaxf(m_run[r], best);
            const float alpha = __expf(m_run[r] - mnew);
            m_run[r] = mnew;
            float rsum = 0.f;
            #pragma unroll
            for (int nb = 0; nb < 2; ++nb) {
                const float p = __expf(pv[nb][r] - mnew);
                rsum += p;
                Ps[(wid * 16 + r + hl * 8) * PSTR + nb * 16 + l16] = f2bf(p);
            }
            #pragma unroll
            for (int off = 1; off < 16; off <<= 1)
                rsum += __shfl_xor(rsum, off, 16);
            l_run[r] = l_run[r] * alpha + rsum;
            #pragma unroll
            for (int c = 0; c < 4; ++c) o[c][r] *= alpha;
        }

        // ---- O += P V : P re-read from LDS in A-fragment layout, 4 WMMAs
        BF16x16 ap;
        {
            const unsigned short* pp = Ps + (wid * 16 + l16) * PSTR + hl * 8;
            ap.u[0] = *(const v4u*)pp;
            ap.u[1] = *(const v4u*)(pp + 16);
        }
        #pragma unroll
        for (int nb = 0; nb < 4; ++nb) {
            BF16x16 bv;
            const unsigned short* vp2 = Vt + (nb * 16 + l16) * VSTR + hl * 8;
            bv.u[0] = *(const v4u*)vp2;
            bv.u[1] = *(const v4u*)(vp2 + 16);
            o[nb] = __builtin_amdgcn_wmma_f32_16x16x32_bf16(
                false, ap.v, false, bv.v, (short)0, o[nb], false, false);
        }
    }

    // ---- epilogue: normalize, store bf16 [B,T,H*HD]
    unsigned short* ob = out + (size_t)(b * TSEQ) * 1024 + h * HD;
    #pragma unroll
    for (int nb = 0; nb < 4; ++nb)
        #pragma unroll
        for (int r = 0; r < 8; ++r) {
            const int row = qr0 + r + hl * 8;
            ob[(size_t)row * 1024 + nb * 16 + l16] = f2bf(o[nb][r] / l_run[r]);
        }
}

// ---------------------------------------------------------------------------
extern "C" void kernel_launch(void* const* d_in, const int* in_sizes, int n_in,
                              void* d_out, int out_size, void* d_ws, size_t ws_size,
                              hipStream_t stream) {
    const float* x    = (const float*)d_in[0];   // [2,2048,1024]
    const float* Wqkv = (const float*)d_in[1];   // [1024,3072]
    const float* Wout = (const float*)d_in[2];   // [1024,1024]
    float* out = (float*)d_out;                  // [2,2048,1024]

    const int Mrows = 2 * TSEQ;                  // 4096
    const size_t nX    = (size_t)Mrows * 1024;   // 4,194,304
    const size_t nWqkv = (size_t)1024 * 3072;    // 3,145,728
    const size_t nWout = (size_t)1024 * 1024;    // 1,048,576

    unsigned short* xb    = (unsigned short*)d_ws;            // bf16 x
    unsigned short* wqkvb = xb + nX;                          // bf16 Wqkv
    unsigned short* woutb = wqkvb + nWqkv;                    // bf16 Wout
    unsigned short* qkvb  = woutb + nWout;                    // bf16 qkv  [4096,3072]
    unsigned short* attnb = qkvb + (size_t)Mrows * 3072;      // bf16 attn [4096,1024]

    // 0) one-shot f32 -> bf16 conversions of inputs
    cvt_f32_bf16<<<dim3(nX / 2048),    256, 0, stream>>>(x,    xb);
    cvt_f32_bf16<<<dim3(nWqkv / 2048), 256, 0, stream>>>(Wqkv, wqkvb);
    cvt_f32_bf16<<<dim3(nWout / 2048), 256, 0, stream>>>(Wout, woutb);

    // 1) QKV projection: [4096,1024] x [1024,3072] -> bf16 qkv
    gemm_bf16<true><<<dim3(3072 / BN, Mrows / BM), 256, 0, stream>>>(
        xb, wqkvb, qkvb, Mrows, 3072, 1024);

    // 2) Flash attention with ALiBi + causal softmax -> bf16 attn
    attn_alibi<<<dim3(2 * NH * (TSEQ / 64)), 128, 0, stream>>>(qkvb, attnb);

    // 3) Output projection: [4096,1024] x [1024,1024] -> f32 d_out
    gemm_bf16<false><<<dim3(1024 / BN, Mrows / BM), 256, 0, stream>>>(
        attnb, woutb, out, Mrows, 1024, 1024);
}
